// PNANoTowersLSPELayer_29368986370544
// MI455X (gfx1250) — matrix operand order repaired and checked by reading, not compiled, not run on hardware
//
#include <hip/hip_runtime.h>
#include <hip/hip_bf16.h>
#include <math.h>
#include <float.h>

typedef __attribute__((ext_vector_type(16))) __bf16 v16bf;
typedef __attribute__((ext_vector_type(8)))  float  v8f;

#define NN   50000
#define EE   800000
#define DD   48
#define FTOT (NN * DD)
#define AVG_LOG 2.8332133440562162f

// workspace layout (float indices)
#define OFF_SUMH  0
#define OFF_SUMP  (4 * FTOT)
#define OFF_DEG   (8 * FTOT)
#define OFF_BNS   (8 * FTOT + NN)
#define OFF_BNQ   (8 * FTOT + NN + 48)
#define OFF_SCALE (8 * FTOT + NN + 96)
#define OFF_SHIFT (8 * FTOT + NN + 144)
#define OFF_PW    (8 * FTOT + NN + 192)   // 32B-aligned (byte ofs % 32 == 0)

// packed-weight regions (bf16 element offsets from OFF_PW)
#define PW_PREH  0
#define PW_PREP  (7 * 3 * 512)            // 10752
#define PW_PRETOT (PW_PREP + 4 * 3 * 512) // 16896 bf16 = 33792 B (preH+preP)
#define PW_POSTH PW_PRETOT
#define PW_TOTAL (PW_POSTH + 21 * 3 * 512)

__device__ __forceinline__ v8f wmma_bf16(v16bf a, v16bf b, v8f c) {
  return __builtin_amdgcn_wmma_f32_16x16x32_bf16(false, a, false, b, (short)0, c,
                                                 false, false);
}

// A-fragment local-K for 16-bit A 16x32 (ISA 7.12.2): lane = kh*16+row,
// element j (VGPR j>>1, half j&1): k = (j>=8?16:0) + 8*kh + (j&7)
__device__ __forceinline__ int klA(int j, int kh) {
  return ((j >> 3) << 4) + (kh << 3) + (j & 7);
}

__device__ __forceinline__ void atomicMaxF(float* a, float v) {
  if (v >= 0.f) atomicMax((int*)a, __float_as_int(v));
  else          atomicMin((unsigned int*)a, (unsigned int)__float_as_int(v));
}
__device__ __forceinline__ void atomicMinF(float* a, float v) {
  if (v >= 0.f) atomicMin((int*)a, __float_as_int(v));
  else          atomicMax((unsigned int*)a, (unsigned int)__float_as_int(v));
}

// ---------------- init: zero sums/deg/bn, +/-FLT_MAX for min/max ----------------
__global__ void init_kernel(float* __restrict__ wsf) {
  const long total = 8L * FTOT + NN + 192;
  for (long i = blockIdx.x * (long)blockDim.x + threadIdx.x; i < total;
       i += (long)gridDim.x * blockDim.x) {
    float v = 0.f;
    if ((i >= 2L * FTOT && i < 3L * FTOT) || (i >= 6L * FTOT && i < 7L * FTOT))
      v = -FLT_MAX;                     // max buffers
    else if ((i >= 3L * FTOT && i < 4L * FTOT) || (i >= 7L * FTOT && i < 8L * FTOT))
      v = FLT_MAX;                      // min buffers
    wsf[i] = v;
  }
}

// ------------- prepack: fp32 weights -> bf16 WMMA B-fragment layout -------------
// B 32x16 (16-bit): VGPR j>>1 half j&1 at lane n: col = n&15, k = (j>=8?16:0)+(n<16?0:8)+(j&7)
__global__ void prepack_kernel(const float* __restrict__ Wph,
                               const float* __restrict__ Wpp,
                               const float* __restrict__ Wqh,
                               float* __restrict__ wsf) {
  int id = blockIdx.x * blockDim.x + threadIdx.x;
  if (id >= 96 * 32) return;
  int f = id >> 5, lane = id & 31;
  const float* W; int K, fl, region;
  if (f < 21)      { W = Wph; K = 208; fl = f;      region = PW_PREH; }
  else if (f < 33) { W = Wpp; K = 112; fl = f - 21; region = PW_PREP; }
  else             { W = Wqh; K = 672; fl = f - 33; region = PW_POSTH; }
  int kt = fl / 3, nt = fl % 3;
  __bf16* dstp = (__bf16*)(wsf + OFF_PW) + region + fl * 512 + lane * 16;
  int col  = nt * 16 + (lane & 15);
  int kofs = (lane < 16) ? 0 : 8;
  for (int j = 0; j < 16; ++j) {
    int k = kt * 32 + ((j >> 3) << 4) + kofs + (j & 7);
    dstp[j] = (__bf16)((k < K) ? W[k * DD + col] : 0.f);
  }
}

// -------- edge kernel: gather tile -> WMMA pretrans (h & p) -> atomic agg -------
__global__ __launch_bounds__(128)
void edge_kernel(const float* __restrict__ h, const float* __restrict__ p,
                 const float* __restrict__ e, const int* __restrict__ src,
                 const int* __restrict__ dst, const float* __restrict__ bph,
                 const float* __restrict__ bpp, float* __restrict__ wsf) {
  __shared__ __bf16 Z[4][16][224];                 // 28672 B
  __shared__ __bf16 WB[PW_PRETOT];                 // 33792 B (preH + preP frags)
  __shared__ int SI[4][16], DI[4][16];             //   512 B
  const int tid = threadIdx.x;
  const int wv = tid >> 5, lane = tid & 31;
  const int base = (blockIdx.x * 4 + wv) * 16;

  // stage all pre-transform B fragments into LDS once per block (33792 B)
  {
    const uint4* srcw = (const uint4*)(wsf + OFF_PW);
    uint4* dstw = (uint4*)WB;
    for (int i = tid; i < PW_PRETOT / 8; i += 128) dstw[i] = srcw[i];
  }
  if (lane < 16) {
    int ei = base + lane;
    bool ok = ei < EE;
    SI[wv][lane] = ok ? src[ei] : 0;
    DI[wv][lane] = ok ? dst[ei] : 0;
  }
  __syncthreads();

  // gather z_h = [h[s], p[s], h[t], p[t], e] (208 cols, pad to 224) as bf16
  for (int r = 0; r < 16; ++r) {
    int ei = base + r;
    bool ok = ei < EE;
    int s = SI[wv][r], t = DI[wv][r];
    for (int c = lane; c < 224; c += 32) {
      float v = 0.f;
      if (ok) {
        if (c < 48)       v = h[s * 48 + c];
        else if (c < 96)  v = p[s * 48 + c - 48];
        else if (c < 144) v = h[t * 48 + c - 96];
        else if (c < 192) v = p[t * 48 + c - 144];
        else if (c < 208) v = e[ei * 16 + c - 192];
      }
      Z[wv][r][c] = (__bf16)v;
    }
  }
  __syncthreads();

  const int row = lane & 15, kh = lane >> 4;
  v8f ch[3] = {}, cp[3] = {};

  // e_for_h: K = 208 (7 k-tiles of 32) x N = 48 (3 n-tiles)
  #pragma unroll
  for (int kt = 0; kt < 7; ++kt) {
    v16bf a;
    #pragma unroll
    for (int j = 0; j < 16; ++j) a[j] = Z[wv][row][kt * 32 + klA(j, kh)];
    #pragma unroll
    for (int nt = 0; nt < 3; ++nt) {
      v16bf b = *(const v16bf*)(WB + PW_PREH + (kt * 3 + nt) * 512 + lane * 16);
      ch[nt] = wmma_bf16(a, b, ch[nt]);
    }
  }
  // e_for_p: z_p cols are a remap of z_h cols: k<48 -> 48+k ; k>=48 -> 96+k
  // (k in [112,128) maps to zero-padded cols 208..223)
  #pragma unroll
  for (int kt = 0; kt < 4; ++kt) {
    v16bf a;
    #pragma unroll
    for (int j = 0; j < 16; ++j) {
      int k = kt * 32 + klA(j, kh);
      int km = k + (k < 48 ? 48 : 96);
      a[j] = Z[wv][row][km];
    }
    #pragma unroll
    for (int nt = 0; nt < 3; ++nt) {
      v16bf b = *(const v16bf*)(WB + PW_PREP + (kt * 3 + nt) * 512 + lane * 16);
      cp[nt] = wmma_bf16(a, b, cp[nt]);
    }
  }

  // scatter: C layout -> lane holds col (lane&15), rows r (+8 for upper half)
  float* sum_h = wsf + OFF_SUMH;
  float* sum_p = wsf + OFF_SUMP;
  #pragma unroll
  for (int nt = 0; nt < 3; ++nt) {
    int d = nt * 16 + (lane & 15);
    float bh = bph[d], bp = bpp[d];
    #pragma unroll
    for (int r = 0; r < 8; ++r) {
      int rr = ((lane >> 4) << 3) + r;
      int ei = base + rr;
      if (ei >= EE) continue;
      int t = DI[wv][rr];
      int o = t * 48 + d;
      float mh = ch[nt][r] + bh;
      atomicAdd(sum_h + o, mh);
      atomicAdd(sum_h + FTOT + o, mh * mh);
      atomicMaxF(sum_h + 2 * FTOT + o, mh);
      atomicMinF(sum_h + 3 * FTOT + o, mh);
      float mp = cp[nt][r] + bp;
      atomicAdd(sum_p + o, mp);
      atomicAdd(sum_p + FTOT + o, mp * mp);
      atomicMaxF(sum_p + 2 * FTOT + o, mp);
      atomicMinF(sum_p + 3 * FTOT + o, mp);
    }
  }
  if (lane < 16 && base + lane < EE)
    atomicAdd(wsf + OFF_DEG + DI[wv][lane], 1.0f);
}

// ----- node kernel: build h_cat[16 x 672], WMMA post-GEMM, BN partial sums -----
__global__ __launch_bounds__(32)
void node_kernel(const float* __restrict__ h, const float* __restrict__ p,
                 const float* __restrict__ snorm, const float* __restrict__ bqh,
                 float* __restrict__ wsf, float* __restrict__ hpre) {
  __shared__ __bf16 Z[16][672];
  __shared__ float sInv[16], sAmp[16], sAtt[16], sHas[16];
  __shared__ float sbn[48], sbq[48];
  const int lane = threadIdx.x;
  const int base = blockIdx.x * 16;

  if (lane < 16) {
    int n = base + lane;
    float dg = (n < NN) ? wsf[OFF_DEG + n] : 0.f;
    float safe = fmaxf(dg, 1.f);
    sInv[lane] = 1.f / safe;
    float logd = logf(dg + 1.f);
    sAmp[lane] = logd / AVG_LOG;
    sAtt[lane] = AVG_LOG / (logd > 0.f ? logd : 1.f);
    sHas[lane] = (dg > 0.f) ? 1.f : 0.f;
  }
  for (int i = lane; i < 48; i += 32) { sbn[i] = 0.f; sbq[i] = 0.f; }
  __syncthreads();

  const float* sum_h = wsf + OFF_SUMH;
  for (int r = 0; r < 16; ++r) {
    int n = base + r;
    bool ok = n < NN;
    float inv = sInv[r], amp = sAmp[r], att = sAtt[r], has = sHas[r];
    for (int c = lane; c < 672; c += 32) {
      float v = 0.f;
      if (ok) {
        if (c < 48)      v = h[n * 48 + c];
        else if (c < 96) v = p[n * 48 + c - 48];
        else {
          int j = c - 96, g = j / 48, d = j - g * 48;
          int o = n * 48 + d;
          int bsel = g & 3, ssel = g >> 2;
          float val;
          if (bsel == 0)      val = sum_h[o] * inv;                 // mean
          else if (bsel == 1) val = sum_h[2 * FTOT + o];            // max
          else if (bsel == 2) val = sum_h[3 * FTOT + o];            // min
          else {                                                    // std
            float mean = sum_h[o] * inv;
            float msq  = sum_h[FTOT + o] * inv;
            val = sqrtf(fmaxf(msq - mean * mean, 0.f) + 1e-5f);
          }
          float sc = (ssel == 0) ? 1.f : (ssel == 1 ? amp : att);
          v = val * sc * has;
        }
      }
      Z[r][c] = (__bf16)v;
    }
  }
  __syncthreads();

  const __bf16* pw = (const __bf16*)(wsf + OFF_PW);
  const int row = lane & 15, kh = lane >> 4;
  v8f acc[3] = {};
  #pragma unroll
  for (int kt = 0; kt < 21; ++kt) {
    v16bf a;
    #pragma unroll
    for (int j = 0; j < 16; ++j) a[j] = Z[row][kt * 32 + klA(j, kh)];
    #pragma unroll
    for (int nt = 0; nt < 3; ++nt) {
      v16bf b = *(const v16bf*)(pw + PW_POSTH + (kt * 3 + nt) * 512 + lane * 16);
      acc[nt] = wmma_bf16(a, b, acc[nt]);
    }
  }

  #pragma unroll
  for (int nt = 0; nt < 3; ++nt) {
    int d = nt * 16 + (lane & 15);
    float bb = bqh[d];
    #pragma unroll
    for (int r = 0; r < 8; ++r) {
      int rr = ((lane >> 4) << 3) + r;
      int n = base + rr;
      if (n >= NN) continue;
      float v = (acc[nt][r] + bb) * snorm[n];   // graph norm
      hpre[n * 48 + d] = v;
      atomicAdd(&sbn[d], v);
      atomicAdd(&sbq[d], v * v);
    }
  }
  __syncthreads();
  for (int i = lane; i < 48; i += 32) {
    atomicAdd(wsf + OFF_BNS + i, sbn[i]);
    atomicAdd(wsf + OFF_BNQ + i, sbq[i]);
  }
}

// ----------------------- BN stats -> affine scale/shift ------------------------
__global__ void bnfinal_kernel(const float* __restrict__ gamma,
                               const float* __restrict__ beta,
                               float* __restrict__ wsf) {
  int d = threadIdx.x;
  if (d < 48) {
    float mu  = wsf[OFF_BNS + d] / (float)NN;
    float var = wsf[OFF_BNQ + d] / (float)NN - mu * mu;
    float sc  = gamma[d] * rsqrtf(var + 1e-5f);
    wsf[OFF_SCALE + d] = sc;
    wsf[OFF_SHIFT + d] = beta[d] - mu * sc;
  }
}

// -------------------- epilogue: BN + relu + residuals + tanh -------------------
__global__ void final_kernel(const float* __restrict__ h, const float* __restrict__ p,
                             const float* __restrict__ wsf, float* __restrict__ dout) {
  int i = blockIdx.x * blockDim.x + threadIdx.x;
  if (i >= FTOT) return;
  int d = i % 48;
  float y = dout[i] * wsf[OFF_SCALE + d] + wsf[OFF_SHIFT + d];
  y = fmaxf(y, 0.f);
  dout[i]        = h[i] + y;          // h_out (overwrites hpre, read-before-write)
  dout[FTOT + i] = p[i] + tanhf(y);   // p_out
}

extern "C" void kernel_launch(void* const* d_in, const int* in_sizes, int n_in,
                              void* d_out, int out_size, void* d_ws, size_t ws_size,
                              hipStream_t stream) {
  const float* h     = (const float*)d_in[0];
  const float* p     = (const float*)d_in[1];
  const float* e     = (const float*)d_in[2];
  const float* snorm = (const float*)d_in[3];
  const int*   src   = (const int*)d_in[4];
  const int*   dst   = (const int*)d_in[5];
  const float* Wph   = (const float*)d_in[6];
  const float* bph   = (const float*)d_in[7];
  const float* Wpp   = (const float*)d_in[8];
  const float* bpp   = (const float*)d_in[9];
  const float* Wqh   = (const float*)d_in[10];
  const float* bqh   = (const float*)d_in[11];
  // d_in[12], d_in[13]: W_post_p / b_post_p — dead in the reference, skipped.
  const float* gamma = (const float*)d_in[14];
  const float* beta  = (const float*)d_in[15];
  float* wsf  = (float*)d_ws;
  float* dout = (float*)d_out;

  hipLaunchKernelGGL(init_kernel,    dim3(2048), dim3(256), 0, stream, wsf);
  hipLaunchKernelGGL(prepack_kernel, dim3(12),   dim3(256), 0, stream, Wph, Wpp, Wqh, wsf);
  hipLaunchKernelGGL(edge_kernel,    dim3((EE + 63) / 64), dim3(128), 0, stream,
                     h, p, e, src, dst, bph, bpp, wsf);
  hipLaunchKernelGGL(node_kernel,    dim3((NN + 15) / 16), dim3(32), 0, stream,
                     h, p, snorm, bqh, wsf, dout);
  hipLaunchKernelGGL(bnfinal_kernel, dim3(1), dim3(64), 0, stream, gamma, beta, wsf);
  hipLaunchKernelGGL(final_kernel,   dim3((FTOT + 255) / 256), dim3(256), 0, stream,
                     h, p, wsf, dout);
}
